// MultiHeadAttentionQuantum_65481071399286
// MI455X (gfx1250) — compile-verified, matrix operand order checked by reading, not run on hardware
//
#include <hip/hip_runtime.h>

// ---------------------------------------------------------------------------
// Quantum multi-head attention, MI455X (gfx1250, wave32, WMMA + async LDS).
//
// Quantum layer closed form (CNOT ladders are linear over GF(2); DEPTH=2
// ladders give M = P^2 with rows {w, w-2, ...}):
//   out_w = prod_{j<=w, (w-j) even} cos(token_j + qparams[0][j] + qparams[1][j])
// ---------------------------------------------------------------------------

typedef __attribute__((ext_vector_type(16))) _Float16 v16h;
typedef __attribute__((ext_vector_type(8)))  _Float16 v8h;
typedef __attribute__((ext_vector_type(4)))  _Float16 v4h;
typedef __attribute__((ext_vector_type(8)))  float    v8f;
typedef int b128_t __attribute__((vector_size(16)));   // matches builtin param

#define EDIM   128
#define DK     8
#define NH     16
#define SEQ    512
#define BATCH  4
#define MROWS  2048            // BATCH*SEQ
#define MT     64              // GEMM M-tile rows per block
#define LDA    136             // LDS stride in halves (16B-aligned runs)
#define LDO    132             // LDS stride (floats) for C staging
#define QT     32              // attention query-tile rows

__device__ __forceinline__ v16h cat16(v8h lo, v8h hi) {
  return __builtin_shufflevector(lo, hi, 0,1,2,3,4,5,6,7,8,9,10,11,12,13,14,15);
}

// Async global->LDS copy, 16 bytes per lane (tracked by ASYNCcnt).
__device__ __forceinline__ void async_ld_b128(const _Float16* g, _Float16* l) {
  __builtin_amdgcn_global_load_async_to_lds_b128(
      (__attribute__((address_space(1))) b128_t*)(uintptr_t)g,
      (__attribute__((address_space(3))) b128_t*)(uintptr_t)l,
      0, 0);
}

// mode: 0 = plain GEMM (out -> outF, fp32)
//       1 = Q (quantum -> qq  f16 [bh][s][d])
//       2 = K (quantum -> kqT f16 [bh][d][s], transposed)
//       3 = V (quantum -> vq  f16 [bh][s][d])
__global__ __launch_bounds__(256) void qgemm_kernel(
    const float* __restrict__ A, const float* __restrict__ W,
    const float* __restrict__ qparams,
    float* __restrict__ outF, _Float16* __restrict__ outH, int mode)
{
  __shared__ _Float16 As[MT * LDA];
  __shared__ _Float16 Ws[EDIM * LDA];
  __shared__ float    Os[MT * LDO];
  __shared__ float    biasS[DK];

  const int t     = threadIdx.x;
  const int Mbase = blockIdx.x * MT;

  __builtin_prefetch(W, 0, 0);   // global_prefetch_b8 into GL2

  if (mode != 0 && t < DK) biasS[t] = qparams[t] + qparams[DK + t];

  // Stage A tile (MT x 128, fp32 -> f16 LDS)
  #pragma unroll
  for (int i = 0; i < 8; ++i) {
    int f = t + i * 256;                      // float4 index
    int row = f >> 5, c4 = f & 31;
    const float4 a4 = *(const float4*)(A + (Mbase + row) * EDIM + c4 * 4);
    v4h h4 = { (_Float16)a4.x, (_Float16)a4.y, (_Float16)a4.z, (_Float16)a4.w };
    *(v4h*)(&As[row * LDA + c4 * 4]) = h4;
  }
  // Stage full weight (128 x 128, fp32 -> f16 LDS)
  #pragma unroll
  for (int i = 0; i < 16; ++i) {
    int f = t + i * 256;
    int row = f >> 5, c4 = f & 31;
    const float4 a4 = *(const float4*)(W + row * EDIM + c4 * 4);
    v4h h4 = { (_Float16)a4.x, (_Float16)a4.y, (_Float16)a4.z, (_Float16)a4.w };
    *(v4h*)(&Ws[row * LDA + c4 * 4]) = h4;
  }
  __syncthreads();

  const int wave  = t >> 5, lane = t & 31;
  const int mrow  = (wave & 3) * 16;   // M sub-tile base (4 tiles of 16)
  const int nhalf = (wave >> 2) * 64;  // N half (2 groups of 4 x 16)
  const int lm    = lane & 15;
  const int g     = lane >> 4;         // lane group for K striping

  v8f acc[4] = {};                     // 4 N-tiles of 16x16 fp32

  #pragma unroll
  for (int kk = 0; kk < 4; ++kk) {     // K = 128 = 4 x 32
    const int kb = kk * 32 + g * 8;
    const _Float16* ap = &As[(mrow + lm) * LDA + kb];
    const v16h afrag = cat16(*(const v8h*)ap, *(const v8h*)(ap + 16));
    #pragma unroll
    for (int nt = 0; nt < 4; ++nt) {
      const _Float16* bp = &Ws[(nhalf + nt * 16 + lm) * LDA + kb];
      const v16h bfrag = cat16(*(const v8h*)bp, *(const v8h*)(bp + 16));
      acc[nt] = __builtin_amdgcn_wmma_f32_16x16x32_f16(
          false, afrag, false, bfrag, (short)0, acc[nt], false, false);
    }
  }

  const int radd = g * 8;              // C layout: VGPR r -> rows r / r+8
  if (mode == 0) {
    #pragma unroll
    for (int nt = 0; nt < 4; ++nt) {
      const int nb = nhalf + nt * 16 + lm;
      #pragma unroll
      for (int r = 0; r < 8; ++r)
        outF[(Mbase + mrow + r + radd) * EDIM + nb] = acc[nt][r];
    }
  } else {
    #pragma unroll
    for (int nt = 0; nt < 4; ++nt) {
      const int nb = nhalf + nt * 16 + lm;
      #pragma unroll
      for (int r = 0; r < 8; ++r)
        Os[(mrow + r + radd) * LDO + nb] = acc[nt][r];
    }
    __syncthreads();
    // Quantum transform: per (row, head) cos-chain products.
    for (int p = t; p < MT * NH; p += 256) {
      const int row = p >> 4, h = p & 15;
      float c[DK], e[DK];
      #pragma unroll
      for (int j = 0; j < DK; ++j)
        c[j] = __cosf(Os[row * LDO + h * DK + j] + biasS[j]);
      e[0] = c[0];        e[1] = c[1];
      e[2] = e[0] * c[2]; e[3] = e[1] * c[3];
      e[4] = e[2] * c[4]; e[5] = e[3] * c[5];
      e[6] = e[4] * c[6]; e[7] = e[5] * c[7];
      const int gm = Mbase + row;
      const int b = gm >> 9, s = gm & (SEQ - 1);
      const int bh = b * NH + h;
      if (mode == 2) {                     // kqT [bh][d][s]
        #pragma unroll
        for (int j = 0; j < DK; ++j)
          outH[(bh * DK + j) * SEQ + s] = (_Float16)e[j];
      } else {                             // qq / vq [bh][s][d]
        #pragma unroll
        for (int j = 0; j < DK; ++j)
          outH[(bh * SEQ + s) * DK + j] = (_Float16)e[j];
      }
    }
  }
}

// One block per (32-query tile, head). K/V tiles DMA'd raw (f16) into LDS via
// async loads (ASYNCcnt), scores materialized in LDS fp32, two-pass
// max-subtracted softmax, then attn @ v.
__global__ __launch_bounds__(256) void qattn_kernel(
    const _Float16* __restrict__ qq, const _Float16* __restrict__ kqT,
    const _Float16* __restrict__ vq, float* __restrict__ outA)
{
  __shared__ _Float16 kh[DK * SEQ];     // 8 KB raw f16, [d][s]
  __shared__ _Float16 vh[SEQ * DK];     // 8 KB raw f16, [s][d]
  __shared__ float sc[QT][SEQ + 4];
  __shared__ float pr[QT][8];
  __shared__ float rmax[QT];
  __shared__ float rsum[QT];

  const int t     = threadIdx.x;
  const int qtile = blockIdx.x;     // 0..15
  const int bh    = blockIdx.y;     // 0..63

  // Async DMA: 4096 halves each, 16B per lane per issue, 2 issues per buffer.
  const _Float16* kb = kqT + bh * DK * SEQ;
  const _Float16* vb = vq  + bh * SEQ * DK;
  {
    const int o = t * 8;                        // halves (16 bytes)
    async_ld_b128(kb + o,        &kh[o]);
    async_ld_b128(kb + o + 2048, &kh[o + 2048]);
    async_ld_b128(vb + o,        &vh[o]);
    async_ld_b128(vb + o + 2048, &vh[o + 2048]);
  }

  const int q   = t >> 3;           // 0..31 query row in tile
  const int seg = t & 7;            // 64-key segment
  float qr[DK];
  const _Float16* qp = qq + (bh * SEQ + qtile * QT + q) * DK;
  #pragma unroll
  for (int j = 0; j < DK; ++j) qr[j] = (float)qp[j];

  asm volatile("s_wait_asynccnt 0x0" ::: "memory");
  __syncthreads();

  const float scale = 0.3535533905932738f;  // 1/sqrt(8)
  float m = -3.0e38f;
  for (int kkk = 0; kkk < 64; ++kkk) {
    const int k = seg * 64 + kkk;
    float s = 0.f;
    #pragma unroll
    for (int j = 0; j < DK; ++j) s = fmaf(qr[j], (float)kh[j * SEQ + k], s);
    s *= scale;
    sc[q][k] = s;
    m = fmaxf(m, s);
  }
  pr[q][seg] = m;
  __syncthreads();
  if (t < QT) {
    float mm = pr[t][0];
    #pragma unroll
    for (int i = 1; i < 8; ++i) mm = fmaxf(mm, pr[t][i]);
    rmax[t] = mm;
  }
  __syncthreads();
  const float mr = rmax[q];
  float ssum = 0.f;
  for (int kkk = 0; kkk < 64; ++kkk) {
    const int k = seg * 64 + kkk;
    const float e = __expf(sc[q][k] - mr);
    sc[q][k] = e;
    ssum += e;
  }
  pr[q][seg] = ssum;
  __syncthreads();
  if (t < QT) {
    float s = 0.f;
    #pragma unroll
    for (int i = 0; i < 8; ++i) s += pr[t][i];
    rsum[t] = s;
  }
  __syncthreads();

  // out[q][d], 256 values, one per thread
  const int qo = t >> 3, d = t & 7;
  float acc = 0.f;
  for (int k = 0; k < SEQ; ++k) acc = fmaf(sc[qo][k], (float)vh[k * DK + d], acc);
  const int b = bh >> 4, h = bh & 15;
  const int srow = qtile * QT + qo;
  outA[(b * SEQ + srow) * EDIM + h * DK + d] = acc / rsum[qo];
}

extern "C" void kernel_launch(void* const* d_in, const int* in_sizes, int n_in,
                              void* d_out, int out_size, void* d_ws, size_t ws_size,
                              hipStream_t stream)
{
  const float* x  = (const float*)d_in[0];
  const float* Wq = (const float*)d_in[1];
  const float* Wk = (const float*)d_in[2];
  const float* Wv = (const float*)d_in[3];
  const float* Wo = (const float*)d_in[4];
  const float* qp = (const float*)d_in[5];

  char* ws = (char*)d_ws;
  _Float16* qq   = (_Float16*)(ws);                    // 512 KB
  _Float16* kqT  = (_Float16*)(ws +  512 * 1024);      // 512 KB
  _Float16* vq   = (_Float16*)(ws + 1024 * 1024);      // 512 KB
  float*    attn = (float*)   (ws + 1536 * 1024);      // 1 MB

  const dim3 gblk(MROWS / MT);   // 32 blocks
  qgemm_kernel<<<gblk, 256, 0, stream>>>(x, Wq, qp, nullptr, qq,  1);
  qgemm_kernel<<<gblk, 256, 0, stream>>>(x, Wk, qp, nullptr, kqT, 2);
  qgemm_kernel<<<gblk, 256, 0, stream>>>(x, Wv, qp, nullptr, vq,  3);
  qattn_kernel<<<dim3(SEQ / QT, BATCH * NH), 256, 0, stream>>>(qq, kqT, vq, attn);
  qgemm_kernel<<<gblk, 256, 0, stream>>>(attn, Wo, qp, (float*)d_out, nullptr, 0);
}
